// HomConv_85744727097473
// MI455X (gfx1250) — matrix operand order repaired and checked by reading, not compiled, not run on hardware
//
#include <hip/hip_runtime.h>
#include <hip/hip_bf16.h>

// ---- CDNA5 vector types for WMMA ----
typedef __attribute__((ext_vector_type(16))) __bf16    v16bf;
typedef __attribute__((ext_vector_type(8)))  float     v8f;
typedef __attribute__((ext_vector_type(8)))  unsigned  v8u;

#define N_NODES_C 100000
#define NUM_HOM_C 500000
#define DIM       128
#define KS        6
#define TILE_M    32          // 2 M-subtiles of 16 per wave

// round-to-nearest-even (weight repack)
__device__ __forceinline__ unsigned short f2bf_rne(float f) {
    unsigned u = __float_as_uint(f);
    u += 0x7FFFu + ((u >> 16) & 1u);
    return (unsigned short)(u >> 16);
}
// cheap round-to-nearest (ties away)
__device__ __forceinline__ unsigned pack2(float lo, float hi) {
    unsigned a = __float_as_uint(lo) + 0x8000u;
    unsigned b = __float_as_uint(hi) + 0x8000u;
    return (a >> 16) | (b & 0xFFFF0000u);
}

// Repack W1/W2 (f32, [k][K][N] row-major) into bf16 fragment-linear layout.
// LDS-staged: coalesced global reads, scattered reads served by LDS.
__global__ __launch_bounds__(256)
void convert_weights(const float* __restrict__ W1,
                     const float* __restrict__ W2,
                     unsigned* __restrict__ wsW) {
    __shared__ float sW[DIM * DIM];                   // 64 KB
    const int km = blockIdx.x;                        // 0..11 = k*2 + m
    const int k  = km >> 1;
    const int m  = km & 1;
    const float* src = (m == 0 ? W1 : W2) + k * DIM * DIM;
    const float4* s4 = (const float4*)src;
    float4* l4 = (float4*)sW;
    for (int i = threadIdx.x; i < DIM * DIM / 4; i += blockDim.x) l4[i] = s4[i];
    __syncthreads();
    unsigned* dst = wsW + km * 8192;
    for (int idx = threadIdx.x; idx < 8192; idx += blockDim.x) {
        int j = idx & 7;
        int L = (idx >> 3) & 31;
        int s = (idx >> 8) & 3;
        int n = (idx >> 10) & 7;
        int N  = 16 * n + (L & 15);
        int K0 = 32 * s + ((j & 4) << 2) + ((L & 16) >> 1) + 2 * (j & 3);
        dst[idx] = (unsigned)f2bf_rne(sW[K0 * DIM + N]) |
                   ((unsigned)f2bf_rne(sW[(K0 + 1) * DIM + N]) << 16);
    }
}

// Pre-convert all of x (f32 row-major) to bf16 packed-pair rows in d_ws.
__global__ void convert_x(const float4* __restrict__ x4,
                          uint2* __restrict__ xbf, int n4) {
    int i = blockIdx.x * blockDim.x + threadIdx.x;
    if (i < n4) {
        float4 p = x4[i];
        xbf[i] = make_uint2(pack2(p.x, p.y), pack2(p.z, p.w));
    }
}

__global__ void zero_out(float4* __restrict__ out, int n4) {
    int i = blockIdx.x * blockDim.x + threadIdx.x;
    if (i < n4) out[i] = make_float4(0.f, 0.f, 0.f, 0.f);
}

template <bool BF16X>
__global__ __launch_bounds__(256)
void homconv_main(const float* __restrict__ x,
                  const unsigned* __restrict__ xbf,
                  const int* __restrict__ mi,
                  const float* __restrict__ b1,
                  const float* __restrict__ b2,
                  const unsigned* __restrict__ wsW,
                  float* __restrict__ out) {
    __shared__ __align__(16) unsigned       s_xg[TILE_M * 64];  // 32x128 bf16 pairs
    __shared__ __align__(16) unsigned short s_h1[TILE_M * DIM]; // 32x128 bf16
    __shared__ int                          s_mi[TILE_M * KS];

    const int t     = threadIdx.x;
    const int wave  = t >> 5;            // 0..7 : N-tile
    const int L     = t & 31;
    const int hom0  = blockIdx.x * TILE_M;
    const int nbase = wave * 16;
    const int lc    = L & 15;            // A-row / B,C-column within tile
    const int hi    = L >> 4;            // lane half
    const int h4    = hi * 4;            // u32 offset for A lane-half K split

    if (t < TILE_M * KS) s_mi[t] = mi[hom0 * KS + t];   // 32x6 idx tile, once

    float bias1s[KS], bias2s[KS];                        // off the critical path
#pragma unroll
    for (int k = 0; k < KS; ++k) {
        bias1s[k] = b1[k * DIM + nbase + lc];
        bias2s[k] = b2[k * DIM + nbase + lc];
    }

    float prod0[8], prod1[8];
#pragma unroll
    for (int r = 0; r < 8; ++r) { prod0[r] = 1.0f; prod1[r] = 1.0f; }

    const unsigned* h1u = (const unsigned*)s_h1;
    __syncthreads();                      // s_mi visible

    for (int k = 0; k < KS; ++k) {
        // ---- gather 32 rows into LDS (bf16 packed pairs) ----
        {
            int row = t >> 3;             // 0..31
            int node = s_mi[row * KS + k];
            if constexpr (BF16X) {        // straight bf16 copy, 32B per thread
                int c32 = (t & 7) * 8;    // u32 offset within row
                const uint4* xp = (const uint4*)(xbf + (size_t)node * 64 + c32);
                uint4* dstp = (uint4*)&s_xg[row * 64 + c32];
                dstp[0] = xp[0];
                dstp[1] = xp[1];
            } else {                      // f32 load + convert
                int c16 = (t & 7) * 16;
                const float4* xp = (const float4*)(x + (size_t)node * DIM + c16);
                float4 p0 = xp[0], p1 = xp[1], p2 = xp[2], p3 = xp[3];
                unsigned* dstp = &s_xg[row * 64 + (c16 >> 1)];
                dstp[0] = pack2(p0.x, p0.y);  dstp[1] = pack2(p0.z, p0.w);
                dstp[2] = pack2(p1.x, p1.y);  dstp[3] = pack2(p1.z, p1.w);
                dstp[4] = pack2(p2.x, p2.y);  dstp[5] = pack2(p2.z, p2.w);
                dstp[6] = pack2(p3.x, p3.y);  dstp[7] = pack2(p3.z, p3.w);
            }
        }

        // ---- preload matmul-1 weight B-fragments (8 global_load_b128) ----
        const unsigned* wb1 = wsW + (((k * 2 + 0) * 8 + wave) * 4) * 256 + L * 8;
        const unsigned* wb2 = wsW + (((k * 2 + 1) * 8 + wave) * 4) * 256 + L * 8;
        v8u bf1[4];
#pragma unroll
        for (int s = 0; s < 4; ++s) bf1[s] = *(const v8u*)(wb1 + s * 256);
        __syncthreads();

        // ---- matmul 1: H1 = relu(Xg @ W1[k] + b1[k]) ; two 16-row subtiles ----
        v8f acc0, acc1;
#pragma unroll
        for (int r = 0; r < 8; ++r) { acc0[r] = bias1s[k]; acc1[r] = bias1s[k]; }
#pragma unroll
        for (int s = 0; s < 4; ++s) {
            const uint4* ap0 = (const uint4*)&s_xg[lc * 64 + 16 * s + h4];
            const uint4* ap1 = (const uint4*)&s_xg[(16 + lc) * 64 + 16 * s + h4];
            uint4 q0 = ap0[0], q1 = ap0[2], q2 = ap1[0], q3 = ap1[2];
            v8u a0 = {q0.x, q0.y, q0.z, q0.w, q1.x, q1.y, q1.z, q1.w};
            v8u a1 = {q2.x, q2.y, q2.z, q2.w, q3.x, q3.y, q3.z, q3.w};
            acc0 = __builtin_amdgcn_wmma_f32_16x16x32_bf16(
                     false, __builtin_bit_cast(v16bf, a0),
                     false, __builtin_bit_cast(v16bf, bf1[s]),
                     (short)0, acc0, false, false);
            acc1 = __builtin_amdgcn_wmma_f32_16x16x32_bf16(
                     false, __builtin_bit_cast(v16bf, a1),
                     false, __builtin_bit_cast(v16bf, bf1[s]),
                     (short)0, acc1, false, false);
        }

        // ---- issue matmul-2 B loads now: overlap relu + LDS + barrier ----
        v8u bf2[4];
#pragma unroll
        for (int s = 0; s < 4; ++s) bf2[s] = *(const v8u*)(wb2 + s * 256);

        // relu + write 32x16 H1 chunk to LDS (bf16 via +0x8000, d16_hi store)
#pragma unroll
        for (int r = 0; r < 8; ++r) {
            unsigned u0 = __float_as_uint(fmaxf(acc0[r], 0.0f)) + 0x8000u;
            unsigned u1 = __float_as_uint(fmaxf(acc1[r], 0.0f)) + 0x8000u;
            int row = hi ? (8 + r) : r;
            s_h1[row * DIM + nbase + lc]        = (unsigned short)(u0 >> 16);
            s_h1[(16 + row) * DIM + nbase + lc] = (unsigned short)(u1 >> 16);
        }
        __syncthreads();

        // ---- matmul 2: O = H1 @ W2[k] + b2[k];  prod *= O ----
        v8f acc2, acc3;
#pragma unroll
        for (int r = 0; r < 8; ++r) { acc2[r] = bias2s[k]; acc3[r] = bias2s[k]; }
#pragma unroll
        for (int s = 0; s < 4; ++s) {
            const uint4* ap0 = (const uint4*)&h1u[lc * 64 + 16 * s + h4];
            const uint4* ap1 = (const uint4*)&h1u[(16 + lc) * 64 + 16 * s + h4];
            uint4 q0 = ap0[0], q1 = ap0[2], q2 = ap1[0], q3 = ap1[2];
            v8u a0 = {q0.x, q0.y, q0.z, q0.w, q1.x, q1.y, q1.z, q1.w};
            v8u a1 = {q2.x, q2.y, q2.z, q2.w, q3.x, q3.y, q3.z, q3.w};
            acc2 = __builtin_amdgcn_wmma_f32_16x16x32_bf16(
                     false, __builtin_bit_cast(v16bf, a0),
                     false, __builtin_bit_cast(v16bf, bf2[s]),
                     (short)0, acc2, false, false);
            acc3 = __builtin_amdgcn_wmma_f32_16x16x32_bf16(
                     false, __builtin_bit_cast(v16bf, a1),
                     false, __builtin_bit_cast(v16bf, bf2[s]),
                     (short)0, acc3, false, false);
        }
#pragma unroll
        for (int r = 0; r < 8; ++r) { prod0[r] *= acc2[r]; prod1[r] *= acc3[r]; }

        if (k + 1 < KS)   // warm next k's weight fragments (global_prefetch_b8)
            __builtin_prefetch(wsW + (((k + 1) * 2 * 8 + wave) * 4) * 256 + L * 8, 0, 1);
        // No end-of-loop barrier: the two in-loop barriers already order this
        // iteration's s_xg/s_h1 reads against next iteration's writes.
    }

    // ---- segment sum: out[mapping_index[h,0]] += prod ----
#pragma unroll
    for (int r = 0; r < 8; ++r) {
        int row = hi ? (8 + r) : r;
        int n0  = s_mi[row * KS];
        int n1  = s_mi[(16 + row) * KS];
        atomicAdd(out + (size_t)n0 * DIM + nbase + lc, prod0[r]);
        atomicAdd(out + (size_t)n1 * DIM + nbase + lc, prod1[r]);
    }
}

extern "C" void kernel_launch(void* const* d_in, const int* in_sizes, int n_in,
                              void* d_out, int out_size, void* d_ws, size_t ws_size,
                              hipStream_t stream) {
    const float* x  = (const float*)d_in[0];
    const int*   mi = (const int*)  d_in[1];
    const float* W1 = (const float*)d_in[2];
    const float* b1 = (const float*)d_in[3];
    const float* W2 = (const float*)d_in[4];
    const float* b2 = (const float*)d_in[5];
    float*    out = (float*)d_out;
    unsigned* wsW = (unsigned*)d_ws;                 // fragments: 384 KB

    convert_weights<<<12, 256, 0, stream>>>(W1, W2, wsW);
    int n4 = out_size / 4;
    zero_out<<<(n4 + 255) / 256, 256, 0, stream>>>((float4*)out, n4);

    const size_t W_BYTES = 12u * 8192u * 4u;                       // 384 KB
    const size_t X_BYTES = (size_t)N_NODES_C * DIM * 2u;           // 25.6 MB
    if (ws_size >= W_BYTES + X_BYTES) {
        unsigned* xbf = wsW + W_BYTES / 4;           // bf16 x, packed pairs
        int xn4 = N_NODES_C * DIM / 4;
        convert_x<<<(xn4 + 255) / 256, 256, 0, stream>>>(
            (const float4*)x, (uint2*)xbf, xn4);
        homconv_main<true><<<NUM_HOM_C / TILE_M, 256, 0, stream>>>(
            x, xbf, mi, b1, b2, wsW, out);
    } else {
        homconv_main<false><<<NUM_HOM_C / TILE_M, 256, 0, stream>>>(
            x, nullptr, mi, b1, b2, wsW, out);
    }
}
